// GeometricGAT_48799418417183
// MI455X (gfx1250) — matrix operand order repaired and checked by reading, not compiled, not run on hardware
//
#include <hip/hip_runtime.h>

// ---------------------------------------------------------------------------
// GeometricGAT (2x GATv2 + projector) for MI455X / gfx1250 (wave32, WMMA).
//  - Dense transforms: v_wmma_f32_16x16x32_bf16, 16x64 tile/wave (4 accums),
//    bf16 pre-converted activations + fragment-packed bf16 weights so every
//    fragment load is a contiguous b128.
//  - Segment softmax/aggregate: ordered-int atomicMax + f32 atomicAdd.
// ---------------------------------------------------------------------------

typedef __attribute__((ext_vector_type(16))) __bf16 v16bf;
typedef __attribute__((ext_vector_type(8)))  __bf16 v8bf;
typedef __attribute__((ext_vector_type(8)))  float  v8f;

#define GAT_N  10000
#define GAT_E  320000
#define GAT_H  4

// Encode float so integer compare == float compare (for atomicMax on int).
__device__ __forceinline__ int enc_f32(float f) {
    int i = __float_as_int(f);
    return (i >= 0) ? i : (i ^ 0x7fffffff);
}
__device__ __forceinline__ float dec_f32(int i) {
    return __int_as_float((i >= 0) ? i : (i ^ 0x7fffffff));
}

// ---------------------------------------------------------------------------
// Init / conversion / repack kernels
// ---------------------------------------------------------------------------
__global__ __launch_bounds__(256) void fill_f32(float* p, float v, int n) {
    int i = blockIdx.x * blockDim.x + threadIdx.x;
    if (i < n) p[i] = v;
}
__global__ __launch_bounds__(256) void fill_i32(int* p, int v, int n) {
    int i = blockIdx.x * blockDim.x + threadIdx.x;
    if (i < n) p[i] = v;
}
__global__ __launch_bounds__(256) void cvt_bf16(const float* __restrict__ in,
                                                __bf16* __restrict__ out, int n) {
    int i = blockIdx.x * blockDim.x + threadIdx.x;
    if (i < n) out[i] = (__bf16)in[i];
}

// Repack W[K,Nout] (f32 row-major) into WMMA-B fragment tiles:
//   Wp[((kb*tilesN + nt)*32 + lane)*16 + j] = bf16( W[kb*32 + (lane/16)*16 + j,
//                                                     nt*16 + (lane%16)] )
// so in the GEMM each lane's 16 B values are contiguous (one 1KB line/wave).
__global__ __launch_bounds__(256)
void repack_w_bf16(const float* __restrict__ W, __bf16* __restrict__ Wp,
                   int K, int Nout) {
    int i = blockIdx.x * blockDim.x + threadIdx.x;
    if (i >= K * Nout) return;
    const int j  = i & 15;
    const int l  = (i >> 4) & 31;
    const int t  = i >> 9;
    const int tilesN = Nout >> 4;
    const int nt = t % tilesN;
    const int kb = t / tilesN;
    const int krow = kb * 32 + (l >> 4) * 16 + j;
    const int ncol = nt * 16 + (l & 15);
    Wp[i] = (__bf16)W[(size_t)krow * Nout + ncol];
}

// ---------------------------------------------------------------------------
// WMMA GEMM:  C[M,Nout] = A_bf16[M,K] @ W_packed + bias[Nout]
// One wave computes a 16(M) x 64(N) tile: 4 accumulators share one A fragment,
// 4 v_wmma_f32_16x16x32_bf16 per K-step of 32.
// A-frag: lane L -> row L%16; elems 0..15: K = kk + (L/16)*16 + j
// C/D   : elem r -> row = r + (L/16)*8, col = L%16  (per 16-wide subtile)
// ---------------------------------------------------------------------------
__global__ __launch_bounds__(256)
void wmma_gemm16x64(const __bf16* __restrict__ A, const __bf16* __restrict__ Wp,
                    const float* __restrict__ bias, float* __restrict__ Co,
                    int M, int K, int Nout) {
    const int lane = threadIdx.x & 31;
    const int wave = threadIdx.x >> 5;
    const int tilesN64 = Nout >> 6;
    const int tile  = blockIdx.x * 8 + wave;
    const int tM    = tile / tilesN64;
    const int tN    = tile - tM * tilesN64;
    if (tM * 16 >= M) return;

    const int half = lane >> 4;
    const int l16  = lane & 15;
    const int rowA = tM * 16 + l16;
    const int tilesN16 = Nout >> 4;

    v8f acc[4] = {};
    for (int kk = 0; kk < K; kk += 32) {
        // A fragment: two contiguous 16B bf16 loads, reused by all 4 WMMAs.
        const __bf16* ap = A + (size_t)rowA * K + kk + half * 16;
        v8bf alo = *(const v8bf*)ap;
        v8bf ahi = *(const v8bf*)(ap + 8);
        v16bf a;
#pragma unroll
        for (int j = 0; j < 8; ++j) { a[j] = alo[j]; a[j + 8] = ahi[j]; }

        const __bf16* wb = Wp + ((size_t)(kk >> 5) * tilesN16 + (tN << 2)) * 512
                              + lane * 16;
#pragma unroll
        for (int s = 0; s < 4; ++s) {
            v16bf b = *(const v16bf*)(wb + (size_t)s * 512);  // contiguous 32B/lane
            acc[s] = __builtin_amdgcn_wmma_f32_16x16x32_bf16(
                /*neg_a=*/false, a, /*neg_b=*/false, b,
                /*c_mod=*/(short)0, acc[s], /*reuse_a=*/false, /*reuse_b=*/false);
        }
    }

#pragma unroll
    for (int s = 0; s < 4; ++s) {
        const int col = (tN << 6) + (s << 4) + l16;
        const float bv = bias ? bias[col] : 0.0f;
#pragma unroll
        for (int r = 0; r < 8; ++r) {
            const int row = (tM << 4) + (half << 3) + r;
            Co[(size_t)row * Nout + col] = acc[s][r] + bv;
        }
    }
}

// ---------------------------------------------------------------------------
// Edge logits: one wave per (edge, head); per-lane contiguous C/32 chunk so
// the unrolled loads coalesce to full-wave b64/b128 clauses.
//   m = leaky_relu(xl[src]+xr[dst]+edge_attr@We, 0.2); logit = <m, att[h]>
// ---------------------------------------------------------------------------
template <int C>
__global__ __launch_bounds__(256)
void edge_logits_t(const float* __restrict__ xl, const float* __restrict__ xr,
                   const float* __restrict__ ea, const float* __restrict__ We,
                   const float* __restrict__ att,
                   const int* __restrict__ src, const int* __restrict__ dst,
                   float* __restrict__ logits, int* __restrict__ lmax_enc) {
    constexpr int HC  = GAT_H * C;
    constexpr int PER = C / 32;
    const int lane = threadIdx.x & 31;
    const int task = blockIdx.x * 8 + (threadIdx.x >> 5);
    if (task >= GAT_E * GAT_H) return;
    const int e = task >> 2, h = task & 3;
    const int s = src[e], d = dst[e];
    const float a0 = ea[e * 3 + 0], a1 = ea[e * 3 + 1], a2 = ea[e * 3 + 2];

    const int cbase = lane * PER;
    const float* xlp = xl + (size_t)s * HC + h * C + cbase;
    const float* xrp = xr + (size_t)d * HC + h * C + cbase;
    const float* w0  = We + h * C + cbase;          // We is [3, H*C] row-major
    const float* atp = att + h * C + cbase;
    __builtin_prefetch(xlp, 0, 3);
    __builtin_prefetch(xrp, 0, 3);

    float sum = 0.0f;
#pragma unroll
    for (int j = 0; j < PER; ++j) {
        const float eev = a0 * w0[j] + a1 * w0[HC + j] + a2 * w0[2 * HC + j];
        float m = xlp[j] + xrp[j] + eev;
        m = (m > 0.0f) ? m : 0.2f * m;
        sum += m * atp[j];
    }
#pragma unroll
    for (int off = 16; off > 0; off >>= 1) sum += __shfl_down(sum, off, 32);
    if (lane == 0) {
        logits[e * GAT_H + h] = sum;
        atomicMax(&lmax_enc[d * GAT_H + h], enc_f32(sum));
    }
}

// ex = exp(logit - segmax[dst]); atomicAdd into denom.  Thread per (e,h).
__global__ __launch_bounds__(256)
void edge_exp(float* __restrict__ logits, const int* __restrict__ lmax_enc,
              float* __restrict__ denom, const int* __restrict__ dst) {
    const int i = blockIdx.x * blockDim.x + threadIdx.x;
    if (i >= GAT_E * GAT_H) return;
    const int e = i >> 2, h = i & 3;
    const int d = dst[e];
    const float ex = __expf(logits[i] - dec_f32(lmax_enc[d * GAT_H + h]));
    logits[i] = ex;
    atomicAdd(&denom[d * GAT_H + h], ex);
}

// alpha = ex/denom[dst]; scatter-add out[dst] += xl[src]*alpha. Wave/(e,h).
template <int C>
__global__ __launch_bounds__(256)
void edge_aggregate_t(const float* __restrict__ xl, const float* __restrict__ ex,
                      const float* __restrict__ denom,
                      const int* __restrict__ src, const int* __restrict__ dst,
                      float* __restrict__ out, float* __restrict__ alpha_out) {
    constexpr int HC  = GAT_H * C;
    constexpr int PER = C / 32;
    const int lane = threadIdx.x & 31;
    const int task = blockIdx.x * 8 + (threadIdx.x >> 5);
    if (task >= GAT_E * GAT_H) return;
    const int e = task >> 2, h = task & 3;
    const int s = src[e], d = dst[e];
    const float alpha = ex[e * GAT_H + h] / denom[d * GAT_H + h];
    if (alpha_out && lane == 0) alpha_out[e * GAT_H + h] = alpha;

    const int cbase = lane * PER;
    const float* xlp = xl + (size_t)s * HC + h * C + cbase;
    float* op        = out + (size_t)d * HC + h * C + cbase;
#pragma unroll
    for (int j = 0; j < PER; ++j) atomicAdd(&op[j], xlp[j] * alpha);
}

// h = silu(agg + bias)   (layer-1 activation)
__global__ __launch_bounds__(256)
void post_bias_silu(float* __restrict__ p, const float* __restrict__ bias, int n, int HC) {
    const int i = blockIdx.x * blockDim.x + threadIdx.x;
    if (i >= n) return;
    const float v = p[i] + bias[i % HC];
    p[i] = v / (1.0f + __expf(-v));
}
// h = agg + bias   (layer-2, no activation before projector)
__global__ __launch_bounds__(256)
void post_bias(float* __restrict__ p, const float* __restrict__ bias, int n, int HC) {
    const int i = blockIdx.x * blockDim.x + threadIdx.x;
    if (i >= n) return;
    p[i] = p[i] + bias[i % HC];
}

// ---------------------------------------------------------------------------
extern "C" void kernel_launch(void* const* d_in, const int* in_sizes, int n_in,
                              void* d_out, int out_size, void* d_ws, size_t ws_size,
                              hipStream_t stream) {
    const float* x    = (const float*)d_in[0];
    const int*   ei   = (const int*)d_in[1];
    const float* ea   = (const float*)d_in[2];
    const float* W1l  = (const float*)d_in[3];
    const float* b1l  = (const float*)d_in[4];
    const float* W1r  = (const float*)d_in[5];
    const float* b1r  = (const float*)d_in[6];
    const float* W1e  = (const float*)d_in[7];
    const float* att1 = (const float*)d_in[8];
    const float* bias1= (const float*)d_in[9];
    const float* W2l  = (const float*)d_in[10];
    const float* b2l  = (const float*)d_in[11];
    const float* W2r  = (const float*)d_in[12];
    const float* b2r  = (const float*)d_in[13];
    const float* W2e  = (const float*)d_in[14];
    const float* att2 = (const float*)d_in[15];
    const float* bias2= (const float*)d_in[16];
    const float* Wp   = (const float*)d_in[17];
    const float* bp   = (const float*)d_in[18];

    const int* src = ei;
    const int* dst = ei + GAT_E;

    float* y_out     = (float*)d_out;                       // [N, 256]
    float* alpha_out = (float*)d_out + (size_t)GAT_N * 256; // [E, 4]

    // ---------------- Workspace carve-up ----------------
    float* ws   = (float*)d_ws;
    float* agg1 = ws;  ws += (size_t)GAT_N * 256;     // layer-1 agg -> h1
    float* big  = ws;  ws += (size_t)GAT_N * 2048;    // xl1/xr1 then xl2/xr2
    float* agg2 = ws;  ws += (size_t)GAT_N * 1024;    // layer-2 agg -> h2
    float* lg   = ws;  ws += (size_t)GAT_E * GAT_H;
    float* den  = ws;  ws += (size_t)GAT_N * GAT_H;
    int*   lmax = (int*)ws;  ws += (size_t)GAT_N * GAT_H;

    float* xl1 = big;
    float* xr1 = big + (size_t)GAT_N * 256;
    float* xl2 = big;                                 // reuses dead xl1/xr1
    float* xr2 = big + (size_t)GAT_N * 1024;

    __bf16* bfw = (__bf16*)ws;
    __bf16* actA_bf = bfw;  bfw += (size_t)GAT_N * 1024;  // x_bf then h2_bf
    __bf16* h1_bf   = bfw;  bfw += (size_t)GAT_N * 256;
    __bf16* W1l_p   = bfw;  bfw += 256 * 256;
    __bf16* W1r_p   = bfw;  bfw += 256 * 256;
    __bf16* W2l_p   = bfw;  bfw += 256 * 1024;
    __bf16* W2r_p   = bfw;  bfw += 256 * 1024;
    __bf16* Wp_p    = bfw;  bfw += 1024 * 256;

    const int EH = GAT_E * GAT_H;
    const int NH = GAT_N * GAT_H;
    const dim3 blk(256);
    auto cdiv = [](int a, int b) { return (a + b - 1) / b; };

    // ---------------- Layer 1 (H=4, C=64) ----------------
    cvt_bf16<<<cdiv(GAT_N * 256, 256), blk, 0, stream>>>(x, actA_bf, GAT_N * 256);
    repack_w_bf16<<<cdiv(256 * 256, 256), blk, 0, stream>>>(W1l, W1l_p, 256, 256);
    repack_w_bf16<<<cdiv(256 * 256, 256), blk, 0, stream>>>(W1r, W1r_p, 256, 256);
    {
        const int tiles = (GAT_N / 16) * (256 / 64);
        wmma_gemm16x64<<<cdiv(tiles, 8), blk, 0, stream>>>(actA_bf, W1l_p, b1l, xl1, GAT_N, 256, 256);
        wmma_gemm16x64<<<cdiv(tiles, 8), blk, 0, stream>>>(actA_bf, W1r_p, b1r, xr1, GAT_N, 256, 256);
    }
    fill_f32<<<cdiv(GAT_N * 256, 256), blk, 0, stream>>>(agg1, 0.0f, GAT_N * 256);
    fill_f32<<<cdiv(NH, 256), blk, 0, stream>>>(den, 0.0f, NH);
    fill_i32<<<cdiv(NH, 256), blk, 0, stream>>>(lmax, (int)0x80000000, NH);

    edge_logits_t<64><<<cdiv(EH, 8), blk, 0, stream>>>(xl1, xr1, ea, W1e, att1, src, dst, lg, lmax);
    edge_exp<<<cdiv(EH, 256), blk, 0, stream>>>(lg, lmax, den, dst);
    edge_aggregate_t<64><<<cdiv(EH, 8), blk, 0, stream>>>(xl1, lg, den, src, dst, agg1, nullptr);
    post_bias_silu<<<cdiv(GAT_N * 256, 256), blk, 0, stream>>>(agg1, bias1, GAT_N * 256, 256);

    // ---------------- Layer 2 (H=4, C=256) ----------------
    cvt_bf16<<<cdiv(GAT_N * 256, 256), blk, 0, stream>>>(agg1, h1_bf, GAT_N * 256);
    repack_w_bf16<<<cdiv(256 * 1024, 256), blk, 0, stream>>>(W2l, W2l_p, 256, 1024);
    repack_w_bf16<<<cdiv(256 * 1024, 256), blk, 0, stream>>>(W2r, W2r_p, 256, 1024);
    {
        const int tiles = (GAT_N / 16) * (1024 / 64);
        wmma_gemm16x64<<<cdiv(tiles, 8), blk, 0, stream>>>(h1_bf, W2l_p, b2l, xl2, GAT_N, 256, 1024);
        wmma_gemm16x64<<<cdiv(tiles, 8), blk, 0, stream>>>(h1_bf, W2r_p, b2r, xr2, GAT_N, 256, 1024);
    }
    fill_f32<<<cdiv(GAT_N * 1024, 256), blk, 0, stream>>>(agg2, 0.0f, GAT_N * 1024);
    fill_f32<<<cdiv(NH, 256), blk, 0, stream>>>(den, 0.0f, NH);
    fill_i32<<<cdiv(NH, 256), blk, 0, stream>>>(lmax, (int)0x80000000, NH);

    edge_logits_t<256><<<cdiv(EH, 8), blk, 0, stream>>>(xl2, xr2, ea, W2e, att2, src, dst, lg, lmax);
    edge_exp<<<cdiv(EH, 256), blk, 0, stream>>>(lg, lmax, den, dst);
    edge_aggregate_t<256><<<cdiv(EH, 8), blk, 0, stream>>>(xl2, lg, den, src, dst, agg2, alpha_out);
    post_bias<<<cdiv(GAT_N * 1024, 256), blk, 0, stream>>>(agg2, bias2, GAT_N * 1024, 1024);

    // ---------------- Projector: y = h2 @ Wp + bp ----------------
    cvt_bf16<<<cdiv(GAT_N * 1024, 256), blk, 0, stream>>>(agg2, actA_bf, GAT_N * 1024);
    repack_w_bf16<<<cdiv(1024 * 256, 256), blk, 0, stream>>>(Wp, Wp_p, 1024, 256);
    {
        const int tiles = (GAT_N / 16) * (256 / 64);
        wmma_gemm16x64<<<cdiv(tiles, 8), blk, 0, stream>>>(actA_bf, Wp_p, bp, y_out, GAT_N, 1024, 256);
    }
}
